// SGCN_29978871726570
// MI455X (gfx1250) — matrix operand order repaired and checked by reading, not compiled, not run on hardware
//
#include <hip/hip_runtime.h>
#include <hip/hip_bf16.h>
#include <math.h>

// ---------- types ----------
typedef __bf16        v16bf __attribute__((ext_vector_type(16)));
typedef float         v8f   __attribute__((ext_vector_type(8)));
typedef unsigned int  v4u   __attribute__((ext_vector_type(4)));
typedef unsigned int  v8u   __attribute__((ext_vector_type(8)));

#define IN_F  128
#define HID   128
#define OUT_F 64
#define K_DIM 128
#define BN_EPS 1e-5f

// ---------------------------------------------------------------------------
// zero fill (float4 granularity; all regions we zero are multiples of 4 floats)
// ---------------------------------------------------------------------------
__global__ __launch_bounds__(256) void zero_f4_kernel(float4* __restrict__ p, int n4) {
    int i = blockIdx.x * blockDim.x + threadIdx.x;
    if (i < n4) p[i] = make_float4(0.f, 0.f, 0.f, 0.f);
}

// ---------------------------------------------------------------------------
// BatchNorm column statistics: partial sums per feature, atomics into ws
// blockDim.x == 128 (one thread per feature), coalesced row-major reads
// ---------------------------------------------------------------------------
__global__ __launch_bounds__(128) void bn_stats_kernel(const float* __restrict__ x,
                                                       float* __restrict__ sums,
                                                       float* __restrict__ sumsq,
                                                       int nrows, int rows_per_block) {
    int f  = threadIdx.x;
    int r0 = blockIdx.x * rows_per_block;
    int r1 = r0 + rows_per_block;
    if (r1 > nrows) r1 = nrows;
    float s = 0.f, q = 0.f;
    for (int r = r0; r < r1; ++r) {
        float v = x[(size_t)r * IN_F + f];
        s += v;
        q += v * v;
    }
    atomicAdd(&sums[f],  s);
    atomicAdd(&sumsq[f], q);
}

__global__ __launch_bounds__(128) void bn_finalize_kernel(const float* __restrict__ sums,
                                                          const float* __restrict__ sumsq,
                                                          const float* __restrict__ gamma,
                                                          const float* __restrict__ beta,
                                                          float* __restrict__ scale,
                                                          float* __restrict__ shift,
                                                          int nrows) {
    int f = threadIdx.x;
    float inv_n = 1.f / (float)nrows;
    float mean = sums[f] * inv_n;
    float var  = sumsq[f] * inv_n - mean * mean;   // biased batch variance
    float sc   = gamma[f] * rsqrtf(var + BN_EPS);
    scale[f] = sc;
    shift[f] = beta[f] - mean * sc;
}

// ---------------------------------------------------------------------------
// Fused (optional BN) + GEMM[ nrows x 128 @ 128 x NCOLS ] + bias + (optional tanh)
// bf16 WMMA path: v_wmma_f32_16x16x32_bf16, fp32 accumulate.
//   - W (K x NCOLS, row-major fp32) staged transposed in LDS as bf16 -> B-fragment
//     is 16 contiguous K values per lane (32B contiguous ds load).
//   - 16-row A tile staged in LDS as bf16 with BN applied -> A-fragment is two
//     contiguous 16B chunks per lane (k = 16*half + s, same k order as B).
//   - One wave per 16-column tile; K=128 -> 4 chained WMMAs.
// ---------------------------------------------------------------------------
template <int NCOLS, bool BN, bool TANH>
__global__ __launch_bounds__(32 * (NCOLS / 16)) void gemm_wmma_kernel(
        const float* __restrict__ A,      // nrows x 128, row-major fp32
        const float* __restrict__ W,      // 128 x NCOLS, row-major fp32
        const float* __restrict__ bias,   // NCOLS
        const float* __restrict__ scale,  // 128 (BN only)
        const float* __restrict__ shift,  // 128 (BN only)
        float* __restrict__ out,          // nrows x NCOLS
        int nrows) {
    constexpr int WAVES = NCOLS / 16;
    constexpr int NT    = 32 * WAVES;

    __shared__ __bf16 sW[NCOLS * K_DIM];   // transposed: sW[n*128 + k]
    __shared__ __bf16 sA[16 * K_DIM];      // row tile:  sA[row*128 + k]

    const int t  = threadIdx.x;
    const int r0 = blockIdx.x * 16;

    // stage W transposed (bf16)
    for (int i = t; i < NCOLS * K_DIM; i += NT) {
        int k = i / NCOLS;
        int n = i % NCOLS;
        sW[n * K_DIM + k] = (__bf16)W[i];
    }
    // stage A tile with BN fused (bf16)
    for (int i = t; i < 16 * K_DIM; i += NT) {
        int row = i >> 7;          // /128
        int c   = i & (K_DIM - 1); // %128
        int gr  = r0 + row;
        float v = (gr < nrows) ? A[(size_t)gr * K_DIM + c] : 0.f;
        if (BN) v = v * scale[c] + shift[c];
        sA[i] = (__bf16)v;
    }
    __syncthreads();

    const int wv   = t >> 5;
    const int lane = t & 31;
    const int c0   = wv * 16;
    const int l15  = lane & 15;
    const int agrp = lane >> 4;    // 0: lanes 0-15, 1: lanes 16-31

    v8f acc = {0.f, 0.f, 0.f, 0.f, 0.f, 0.f, 0.f, 0.f};

#pragma unroll
    for (int kb = 0; kb < K_DIM; kb += 32) {
        // A fragment: row = l15; K = kb+agrp*16 + {0..15}
        const __bf16* pa = &sA[l15 * K_DIM + kb + agrp * 16];
        v4u alo = *(const v4u*)(pa);        // K .. K+7   (16B)
        v4u ahi = *(const v4u*)(pa + 8);    // K+8..K+15  (16B)
        v8u au;
#pragma unroll
        for (int j = 0; j < 4; ++j) { au[j] = alo[j]; au[j + 4] = ahi[j]; }
        v16bf afrag = __builtin_bit_cast(v16bf, au);

        // B fragment: col = c0+l15; K = kb+agrp*16 .. +15 (contiguous 32B)
        const __bf16* pb = &sW[(c0 + l15) * K_DIM + kb + agrp * 16];
        v8u bu = *(const v8u*)(pb);
        v16bf bfrag = __builtin_bit_cast(v16bf, bu);

        acc = __builtin_amdgcn_wmma_f32_16x16x32_bf16(
            /*neg_a=*/false, afrag, /*neg_b=*/false, bfrag,
            /*c_mod=*/(short)0, acc, /*reuse_a=*/false, /*reuse_b=*/false);
    }

    // epilogue: C/D layout -> VGPR v holds M = v + agrp*8, N = lane&15
    const int ocol  = c0 + l15;
    const int orow0 = r0 + agrp * 8;
    const float bv  = bias[ocol];
#pragma unroll
    for (int v = 0; v < 8; ++v) {
        int orow = orow0 + v;
        if (orow < nrows) {
            float val = acc[v] + bv;
            if (TANH) val = tanhf(val);
            out[(size_t)orow * NCOLS + ocol] = val;
        }
    }
}

// ---------------------------------------------------------------------------
// SpMM hop: out[dst] += val * in[src]   (COO edge-parallel, 32 lanes per edge,
// float4 per lane = 128 features). Grid-stride over edge groups with a
// prefetch of the next iteration's gather row (global_prefetch_b8).
// ---------------------------------------------------------------------------
__global__ __launch_bounds__(256) void spmm_kernel(const int* __restrict__ esrc,
                                                   const int* __restrict__ edst,
                                                   const float* __restrict__ eval,
                                                   const float* __restrict__ vin,
                                                   float* __restrict__ vout,
                                                   int nedges, int egroup_stride) {
    int g  = blockIdx.x * blockDim.x + threadIdx.x;
    int e  = g >> 5;             // edge index (32 lanes per edge)
    int f4 = (g & 31) * 4;       // feature offset (float4 per lane)

    for (; e < nedges; e += egroup_stride) {
        int en = e + egroup_stride;
        if (en < nedges) {
            int sn = esrc[en];
            __builtin_prefetch(vin + (size_t)sn * HID + f4, 0, 0);
        }
        int   s = esrc[e];
        int   d = edst[e];
        float w = eval[e];
        float4 v = *(const float4*)(vin + (size_t)s * HID + f4);
        float* o = vout + (size_t)d * HID + f4;
        atomicAdd(o + 0, w * v.x);
        atomicAdd(o + 1, w * v.y);
        atomicAdd(o + 2, w * v.z);
        atomicAdd(o + 3, w * v.w);
    }
}

// ---------------------------------------------------------------------------
// launch
// ---------------------------------------------------------------------------
extern "C" void kernel_launch(void* const* d_in, const int* in_sizes, int n_in,
                              void* d_out, int out_size, void* d_ws, size_t ws_size,
                              hipStream_t stream) {
    const float* x     = (const float*)d_in[0];
    const int*   esrc  = (const int*)  d_in[1];
    const int*   edst  = (const int*)  d_in[2];
    const float* eval  = (const float*)d_in[3];
    const float* gamma = (const float*)d_in[4];
    const float* beta  = (const float*)d_in[5];
    const float* W_in  = (const float*)d_in[6];
    const float* b_in  = (const float*)d_in[7];
    const float* W_sg  = (const float*)d_in[8];
    const float* b_sg  = (const float*)d_in[9];
    const float* W_out = (const float*)d_in[10];
    const float* b_out = (const float*)d_in[11];

    const int N = in_sizes[0] / IN_F;   // 100000
    const int E = in_sizes[1];          // 1600000

    // workspace layout (floats): [scale 128][shift 128][sums 128][sumsq 128]
    // pad to 1024, then hA (N*128), hB (N*128)
    float* wsf   = (float*)d_ws;
    float* scale = wsf;
    float* shift = wsf + 128;
    float* sums  = wsf + 256;
    float* sumsq = wsf + 384;
    float* hA    = wsf + 1024;
    float* hB    = hA + (size_t)N * HID;

    // --- BatchNorm statistics ---
    zero_f4_kernel<<<1, 256, 0, stream>>>((float4*)sums, 64);   // sums+sumsq = 256 floats
    {
        int blocks = 512;
        int rows_per = (N + blocks - 1) / blocks;
        bn_stats_kernel<<<blocks, 128, 0, stream>>>(x, sums, sumsq, N, rows_per);
    }
    bn_finalize_kernel<<<1, 128, 0, stream>>>(sums, sumsq, gamma, beta, scale, shift, N);

    const int gemm_blocks = (N + 15) / 16;

    // --- h = tanh(BN(x) @ W_in + b_in) -> hA ---
    gemm_wmma_kernel<HID, true, true><<<gemm_blocks, 256, 0, stream>>>(
        x, W_in, b_in, scale, shift, hA, N);

    // --- 4 SpMM hops, double-buffered hA <-> hB ---
    {
        const int n4h = N * HID / 4;
        const int zblocks = (n4h + 255) / 256;
        const int sblocks = 100000;                         // grid-stride: ~2 trips
        const int egroup_stride = sblocks * 256 / 32;       // edge slots per pass
        float* cur = hA;
        float* nxt = hB;
        for (int hop = 0; hop < 4; ++hop) {
            zero_f4_kernel<<<zblocks, 256, 0, stream>>>((float4*)nxt, n4h);
            spmm_kernel<<<sblocks, 256, 0, stream>>>(esrc, edst, eval, cur, nxt,
                                                     E, egroup_stride);
            float* tmp = cur; cur = nxt; nxt = tmp;
        }
        // after 4 hops result is back in hA (cur == hA)
    }

    // --- h = tanh(h @ W_sg + b_sg) -> hB ---
    gemm_wmma_kernel<HID, false, true><<<gemm_blocks, 256, 0, stream>>>(
        hA, W_sg, b_sg, nullptr, nullptr, hB, N);

    // --- out = h @ W_out + b_out -> d_out ---
    gemm_wmma_kernel<OUT_F, false, false><<<gemm_blocks, 128, 0, stream>>>(
        hB, W_out, b_out, nullptr, nullptr, (float*)d_out, N);
}
